// Structure_40346922778720
// MI455X (gfx1250) — matrix operand order repaired and checked by reading, not compiled, not run on hardware
//
#include <hip/hip_runtime.h>
#include <hip/hip_bf16.h>
#include <math.h>

typedef _Float16 f16;
typedef long long ll;
typedef __attribute__((ext_vector_type(16))) _Float16 v16h;
typedef __attribute__((ext_vector_type(8)))  _Float16 v8h;
typedef __attribute__((ext_vector_type(8)))  float    v8f;
typedef int v4i_ __attribute__((vector_size(16)));

#ifndef __has_builtin
#define __has_builtin(x) 0
#endif
#if __has_builtin(__builtin_amdgcn_global_load_async_to_lds_b128) && \
    __has_builtin(__builtin_amdgcn_s_wait_asynccnt)
#define USE_ASYNC_LDS 1
#else
#define USE_ASYNC_LDS 0
#endif

// ---------------------------------------------------------------------------
// WMMA GEMM:  C[M,N] = sum_k A[m,k] * B[n,k]   (A:[MxK] f16, B:[NxK] f16, C f32)
// K must be a multiple of 32 (callers pad). Block: 256 thr = 8 waves,
// tile 128(M) x 64(N); each wave computes a 32x32 region as 2x2 WMMA tiles.
// Double-buffered LDS pipeline: async-copy tile k+1 overlaps WMMA on tile k.
// ---------------------------------------------------------------------------
__device__ __forceinline__ void stage_tiles(
    const f16* __restrict__ A, const f16* __restrict__ B, int M, int N, int K,
    int m0, int n0, int k0, int tid, f16* As, f16* Bs) {
  { // A tile: 128 rows x 32 f16; one half-row (16 f16 = 32B) per thread
    const int row  = tid >> 1;
    const int half = tid & 1;
    const int gr   = m0 + row;
    f16* dst = As + row * 32 + half * 16;
    if (gr < M) {
      const f16* src = A + (ll)gr * K + k0 + half * 16;
#if USE_ASYNC_LDS
      __builtin_amdgcn_global_load_async_to_lds_b128(
          (__attribute__((address_space(1))) v4i_*)(unsigned long long)(const void*)src,
          (__attribute__((address_space(3))) v4i_*)(unsigned)(unsigned long long)(void*)dst,
          0, 0);
      __builtin_amdgcn_global_load_async_to_lds_b128(
          (__attribute__((address_space(1))) v4i_*)(unsigned long long)(const void*)(src + 8),
          (__attribute__((address_space(3))) v4i_*)(unsigned)(unsigned long long)(void*)(dst + 8),
          0, 0);
#else
      *(v8h*)dst       = *(const v8h*)src;
      *(v8h*)(dst + 8) = *(const v8h*)(src + 8);
#endif
    } else {
      v8h z = {};
      *(v8h*)dst = z; *(v8h*)(dst + 8) = z;
    }
  }
  { // B tile (n-major): 64 rows x 32 f16; 8 f16 (16B) per thread
    const int row = tid >> 2;
    const int q   = tid & 3;
    const int gn  = n0 + row;
    f16* dst = Bs + row * 32 + q * 8;
    if (gn < N) {
      const f16* src = B + (ll)gn * K + k0 + q * 8;
#if USE_ASYNC_LDS
      __builtin_amdgcn_global_load_async_to_lds_b128(
          (__attribute__((address_space(1))) v4i_*)(unsigned long long)(const void*)src,
          (__attribute__((address_space(3))) v4i_*)(unsigned)(unsigned long long)(void*)dst,
          0, 0);
#else
      *(v8h*)dst = *(const v8h*)src;
#endif
    } else {
      v8h z = {};
      *(v8h*)dst = z;
    }
  }
}

__global__ __launch_bounds__(256) void k_wmma_gemm(
    const f16* __restrict__ A, const f16* __restrict__ B, float* __restrict__ C,
    int M, int N, int K) {
  __shared__ __align__(16) f16 As[2][128 * 32];
  __shared__ __align__(16) f16 Bs[2][64 * 32];
  const int tid  = threadIdx.x;
  const int lane = tid & 31;
  const int wave = tid >> 5;
  const int wy = wave >> 1;   // 0..3
  const int wx = wave & 1;    // 0..1
  const int m0 = blockIdx.y * 128;
  const int n0 = blockIdx.x * 64;
  const int lm = lane & 15;   // row/col within fragment
  const int lh = lane >> 4;   // half select
  v8f acc[2][2] = {};
  // Prologue: stage tile 0 into buffer 0.
  stage_tiles(A, B, M, N, K, m0, n0, 0, tid, As[0], Bs[0]);
  for (int k0 = 0; k0 < K; k0 += 32) {
    const int buf = (k0 >> 5) & 1;
#if USE_ASYNC_LDS
    __builtin_amdgcn_s_wait_asynccnt(0);   // this wave's copy of tile k0 landed
#endif
    // Barrier proves (a) tile k0 staged by all waves, (b) everyone finished
    // computing on the buffer we are about to restage below.
    __syncthreads();
    if (k0 + 32 < K) {
      stage_tiles(A, B, M, N, K, m0, n0, k0 + 32, tid, As[buf ^ 1], Bs[buf ^ 1]);
      __builtin_prefetch(A + (ll)(m0 + (tid >> 1)) * K + k0 + 64, 0, 1);
    }
    // Compute on tile k0 while tile k0+32 streams into the other buffer.
    union ABu { v16h v; v8h h[2]; };
    ABu a[2];
    const f16* Ab = As[buf];
    const f16* Bb = Bs[buf];
    for (int i = 0; i < 2; ++i) {
      const int am = wy * 32 + i * 16 + lm;
      a[i].h[0] = *(const v8h*)&Ab[am * 32 + lh * 8];
      a[i].h[1] = *(const v8h*)&Ab[am * 32 + 16 + lh * 8];
    }
    for (int j = 0; j < 2; ++j) {
      ABu b;
      const int bn = wx * 32 + j * 16 + lm;
      b.h[0] = *(const v8h*)&Bb[bn * 32 + lh * 8];
      b.h[1] = *(const v8h*)&Bb[bn * 32 + 16 + lh * 8];
      for (int i = 0; i < 2; ++i) {
        acc[i][j] = __builtin_amdgcn_wmma_f32_16x16x32_f16(
            false, a[i].v, false, b.v, (short)0, acc[i][j], false, false);
      }
    }
  }
  for (int i = 0; i < 2; ++i)
    for (int j = 0; j < 2; ++j) {
      const int nn = n0 + wx * 32 + j * 16 + lm;
      if (nn >= N) continue;
      for (int r = 0; r < 8; ++r) {
        const int mm = m0 + wy * 32 + i * 16 + lh * 8 + r;
        if (mm < M) C[(ll)mm * N + nn] = acc[i][j][r];
      }
    }
}

// ---------------------------------------------------------------------------
// Elementwise / data movement kernels
// ---------------------------------------------------------------------------
__global__ void k_prep(const float* __restrict__ x, float* __restrict__ o, ll n) {
  ll i = (ll)blockIdx.x * blockDim.x + threadIdx.x;
  if (i < n) o[i] = 2.0f * (x[i] * (1.0f / 255.0f)) - 1.0f;
}

__global__ void k_cast_pad(const float* __restrict__ src, int srcld, int col0, int cols,
                           f16* __restrict__ dst, int dstld, ll n) {
  ll i = (ll)blockIdx.x * blockDim.x + threadIdx.x;
  if (i >= n) return;
  ll row = i / dstld; int c = (int)(i % dstld);
  dst[i] = (c < cols) ? (f16)src[row * srcld + col0 + c] : (f16)0.0f;
}

__global__ void k_im2col(const float* __restrict__ src, f16* __restrict__ dst,
                         int N, int Cin, int H, int W, int OH, int OW,
                         int KH, int KW, int stride, int pad, int Kc, int Kp, ll total) {
  ll i = (ll)blockIdx.x * blockDim.x + threadIdx.x;
  if (i >= total) return;
  ll m = i / Kp; int kk = (int)(i % Kp);
  f16 val = (f16)0.0f;
  if (kk < Kc) {
    int ow = (int)(m % OW); int oh = (int)((m / OW) % OH); int n = (int)(m / ((ll)OW * OH));
    int kw = kk % KW; int kh = (kk / KW) % KH; int ci = kk / (KW * KH);
    int ih = oh * stride - pad + kh;
    int iw = ow * stride - pad + kw;
    if (ih >= 0 && ih < H && iw >= 0 && iw < W)
      val = (f16)src[((ll)(n * Cin + ci) * H + ih) * W + iw];
  }
  dst[i] = val;
}

__global__ void k_bn_act(const float* __restrict__ go, float* __restrict__ out,
                         const float* g, const float* b, const float* m, const float* v,
                         const float* __restrict__ resid, int relu,
                         int N, int C, int OH, int OW, ll total) {
  ll i = (ll)blockIdx.x * blockDim.x + threadIdx.x;
  if (i >= total) return;
  ll hw = (ll)OH * OW;
  int p = (int)(i % hw);
  int c = (int)((i / hw) % C);
  int n = (int)(i / (hw * C));
  ll row = (ll)n * hw + p;
  float s = g[c] * rsqrtf(v[c] + 1e-5f);
  float val = go[row * C + c] * s + (b[c] - m[c] * s);
  if (resid) val += resid[i];
  if (relu) val = fmaxf(val, 0.0f);
  out[i] = val;
}

__global__ void k_maxpool(const float* __restrict__ in, float* __restrict__ out,
                          int N, int C, int H, int W, int OH, int OW, ll total) {
  ll i = (ll)blockIdx.x * blockDim.x + threadIdx.x;
  if (i >= total) return;
  int ow = (int)(i % OW); int oh = (int)((i / OW) % OH);
  int c  = (int)((i / ((ll)OW * OH)) % C);
  int n  = (int)(i / ((ll)OW * OH * C));
  float mx = -3.4e38f;
  for (int kh = 0; kh < 3; ++kh)
    for (int kw = 0; kw < 3; ++kw) {
      int ih = oh * 2 - 1 + kh, iw = ow * 2 - 1 + kw;
      if (ih >= 0 && ih < H && iw >= 0 && iw < W)
        mx = fmaxf(mx, in[((ll)(n * C + c) * H + ih) * W + iw]);
    }
  out[i] = mx;
}

__global__ void k_add(float* __restrict__ a, const float* __restrict__ b, ll n) {
  ll i = (ll)blockIdx.x * blockDim.x + threadIdx.x;
  if (i < n) a[i] += b[i];
}

__global__ void k_bias_act(float* __restrict__ x, const float* __restrict__ b,
                           int N, int relu, ll total) {
  ll i = (ll)blockIdx.x * blockDim.x + threadIdx.x;
  if (i >= total) return;
  float v = x[i] + b[i % N];
  if (relu) v = fmaxf(v, 0.0f);
  x[i] = v;
}

__global__ void k_copy_cols(const float* __restrict__ src, int srcld,
                            float* __restrict__ dst, int dstld, int col0, ll total) {
  ll i = (ll)blockIdx.x * blockDim.x + threadIdx.x;
  if (i >= total) return;
  ll row = i / srcld; int c = (int)(i % srcld);
  dst[row * dstld + col0 + c] = src[i];
}

__global__ void k_avg_hw(const float* __restrict__ act, float* __restrict__ out,
                         int C, int S, ll total) {
  ll i = (ll)blockIdx.x * blockDim.x + threadIdx.x;
  if (i >= total) return;
  const float* p = act + i * (ll)S;
  float s = 0.0f;
  for (int k = 0; k < S; ++k) s += p[k];
  out[i] = s / (float)S;
}

__global__ void k_interp(const float* __restrict__ src, int Lr, int Cin,
                         float* __restrict__ dst, int Cout, ll total) {
  ll i = (ll)blockIdx.x * blockDim.x + threadIdx.x;
  if (i >= total) return;
  int c = (int)(i % Cout); ll l = i / Cout;
  float pos = (Cout > 1) ? c * ((float)(Cin - 1) / (float)(Cout - 1)) : 0.0f;
  int lo = (int)floorf(pos); if (lo < 0) lo = 0; if (lo > Cin - 1) lo = Cin - 1;
  int hi = lo + 1; if (hi > Cin - 1) hi = Cin - 1;
  float w = pos - (float)lo;
  dst[i] = src[l * Cin + lo] * (1.0f - w) + src[l * Cin + hi] * w;
}

__global__ void k_nchw_to_tok(const float* __restrict__ act, float* __restrict__ tok,
                              int Lseq, int C, int S, ll total) {
  ll i = (ll)blockIdx.x * blockDim.x + threadIdx.x;
  if (i >= total) return;
  int c = (int)(i % C); ll t = i / C;
  int l = (int)(t % Lseq); ll s = t / Lseq;
  tok[i] = act[((ll)(l * C + c)) * S + s];
}

__global__ void k_tok_add_nchw(float* __restrict__ act, const float* __restrict__ tok,
                               int Lseq, int C, int S, ll total) {
  ll i = (ll)blockIdx.x * blockDim.x + threadIdx.x;
  if (i >= total) return;
  int c = (int)(i % C); ll t = i / C;
  int l = (int)(t % Lseq); ll s = t / Lseq;
  act[((ll)(l * C + c)) * S + s] += tok[i];
}

// ---------------------------------------------------------------------------
// Mamba pieces
// ---------------------------------------------------------------------------
__global__ void k_dwconv_silu(const float* __restrict__ xz, const float* __restrict__ w,
                              const float* __restrict__ b, float* __restrict__ xc,
                              f16* __restrict__ xc16, int S, int Lseq, int di, ll total) {
  ll i = (ll)blockIdx.x * blockDim.x + threadIdx.x;
  if (i >= total) return;
  int d = (int)(i % di); ll t = i / di;
  int l = (int)(t % Lseq); ll s = t / Lseq;
  float acc = b[d];
  for (int j = 0; j < 4; ++j) {
    int lp = l - 3 + j;
    if (lp >= 0) acc += w[d * 4 + j] * xz[(s * Lseq + lp) * (ll)(2 * di) + d];
  }
  float sv = acc * (1.0f / (1.0f + __expf(-acc)));
  xc[i] = sv; xc16[i] = (f16)sv;
}

__global__ void k_softplus_bias(float* __restrict__ x, const float* __restrict__ b,
                                int di, ll total) {
  ll i = (ll)blockIdx.x * blockDim.x + threadIdx.x;
  if (i >= total) return;
  float v = x[i] + b[i % di];
  x[i] = (v > 20.0f) ? v : __logf(1.0f + __expf(v));
}

__global__ void k_scan(const float* __restrict__ dt, const float* __restrict__ dbl,
                       int rn, int r, const float* __restrict__ xc,
                       const float* __restrict__ xz, const float* __restrict__ A_log,
                       const float* __restrict__ Dp, float* __restrict__ y,
                       f16* __restrict__ y16, int Lseq, int di, ll total) {
  ll i = (ll)blockIdx.x * blockDim.x + threadIdx.x;
  if (i >= total) return;           // total = S*di
  int d = (int)(i % di); ll s = i / di;
  float An[16], h[16];
  for (int n = 0; n < 16; ++n) { An[n] = -__expf(A_log[d * 16 + n]); h[n] = 0.0f; }
  for (int l = 0; l < Lseq; ++l) {
    ll t = s * Lseq + l;
    float dtv = dt[t * di + d];
    float xv  = xc[t * di + d];
    const float* bl = dbl + t * rn + r;
    float acc = 0.0f;
    for (int n = 0; n < 16; ++n) {
      float da = __expf(dtv * An[n]);
      h[n] = da * h[n] + dtv * bl[n] * xv;
      acc += h[n] * bl[16 + n];
    }
    acc += xv * Dp[d];
    float zv = xz[t * (ll)(2 * di) + di + d];
    float o  = acc * (zv * (1.0f / (1.0f + __expf(-zv))));
    y[t * di + d] = o; y16[t * di + d] = (f16)o;
  }
}

// ---------------------------------------------------------------------------
// Fuser pieces
// ---------------------------------------------------------------------------
__global__ void k_layernorm(const float* __restrict__ in, const float* __restrict__ g,
                            const float* __restrict__ b, float* __restrict__ out, int C) {
  int row = blockIdx.x, tid = threadIdx.x;
  __shared__ float red[256];
  const float* p = in + (ll)row * C;
  float s = 0.0f;
  for (int c = tid; c < C; c += 256) s += p[c];
  red[tid] = s; __syncthreads();
  for (int o = 128; o > 0; o >>= 1) { if (tid < o) red[tid] += red[tid + o]; __syncthreads(); }
  float mean = red[0] / (float)C; __syncthreads();
  float vs = 0.0f;
  for (int c = tid; c < C; c += 256) { float d = p[c] - mean; vs += d * d; }
  red[tid] = vs; __syncthreads();
  for (int o = 128; o > 0; o >>= 1) { if (tid < o) red[tid] += red[tid + o]; __syncthreads(); }
  float inv = rsqrtf(red[0] / (float)C + 1e-5f); __syncthreads();
  for (int c = tid; c < C; c += 256)
    out[(ll)row * C + c] = (p[c] - mean) * inv * g[c] + b[c];
}

__global__ void k_cross_attn(const float* __restrict__ xs, const float* __restrict__ sens,
                             const float* __restrict__ wK, const float* __restrict__ bK,
                             const float* __restrict__ wV, const float* __restrict__ bV,
                             float* __restrict__ emb, int C) {
  int l = blockIdx.x, tid = threadIdx.x;
  __shared__ float red[256];
  __shared__ float dots[4];
  float sv[4][3];
  for (int k = 0; k < 4; ++k)
    for (int i = 0; i < 3; ++i) sv[k][i] = sens[l * 12 + k * 3 + i];
  float pd[4] = {0.0f, 0.0f, 0.0f, 0.0f};
  for (int c = tid; c < C; c += 256) {
    float xv = xs[(ll)l * C + c];
    for (int k = 0; k < 4; ++k) {
      float kk = bK[k * C + c];
      for (int i = 0; i < 3; ++i) kk += sv[k][i] * wK[((ll)(k * C + c)) * 3 + i];
      pd[k] += kk * xv;
    }
  }
  for (int k = 0; k < 4; ++k) {
    red[tid] = pd[k]; __syncthreads();
    for (int o = 128; o > 0; o >>= 1) { if (tid < o) red[tid] += red[tid + o]; __syncthreads(); }
    if (tid == 0) dots[k] = red[0];
    __syncthreads();
  }
  float scale = rsqrtf((float)C);
  float t0 = dots[0] * scale, t1 = dots[1] * scale, t2 = dots[2] * scale, t3 = dots[3] * scale;
  float mx = fmaxf(fmaxf(t0, t1), fmaxf(t2, t3));
  float e0 = __expf(t0 - mx), e1 = __expf(t1 - mx), e2 = __expf(t2 - mx), e3 = __expf(t3 - mx);
  float inv = 1.0f / (e0 + e1 + e2 + e3);
  float sc[4] = {e0 * inv, e1 * inv, e2 * inv, e3 * inv};
  for (int c = tid; c < C; c += 256) {
    float acc = 0.0f;
    for (int k = 0; k < 4; ++k) {
      float vv = bV[k * C + c];
      for (int i = 0; i < 3; ++i) vv += sv[k][i] * wV[((ll)(k * C + c)) * 3 + i];
      acc += sc[k] * vv;
    }
    emb[(ll)l * C + c] = acc;
  }
}

__global__ void k_attn_scores(const float* __restrict__ qh, const float* __restrict__ kh,
                              float* __restrict__ att, int Lq, int nh, int hd,
                              float scale, ll total) {
  ll i = (ll)blockIdx.x * blockDim.x + threadIdx.x;
  if (i >= total) return;
  int m = (int)(i % Lq); int h = (int)((i / Lq) % nh); int l = (int)(i / ((ll)Lq * nh));
  const float* q = qh + (ll)l * nh * hd + h * hd;
  const float* k = kh + (ll)m * nh * hd + h * hd;
  float s = 0.0f;
  for (int d = 0; d < hd; ++d) s += q[d] * k[d];
  att[i] = s * scale;
}

__global__ void k_softmax_rows(float* __restrict__ x, int rows, int n) {
  int r = blockIdx.x * blockDim.x + threadIdx.x;
  if (r >= rows) return;
  float* p = x + (ll)r * n;
  float mx = p[0];
  for (int i = 1; i < n; ++i) mx = fmaxf(mx, p[i]);
  float s = 0.0f;
  for (int i = 0; i < n; ++i) { float e = __expf(p[i] - mx); p[i] = e; s += e; }
  float inv = 1.0f / s;
  for (int i = 0; i < n; ++i) p[i] *= inv;
}

__global__ void k_attn_out(const float* __restrict__ att, const float* __restrict__ vh,
                           float* __restrict__ o, int Lq, int nh, int hd, ll total) {
  ll i = (ll)blockIdx.x * blockDim.x + threadIdx.x;
  if (i >= total) return;
  int C = nh * hd;
  int c = (int)(i % C); int l = (int)(i / C);
  int h = c / hd;
  float s = 0.0f;
  for (int m = 0; m < Lq; ++m) s += att[((ll)l * nh + h) * Lq + m] * vh[(ll)m * C + c];
  o[i] = s;
}

// ---------------------------------------------------------------------------
// Host-side orchestration
// ---------------------------------------------------------------------------
namespace {

struct Ws {
  char* base; size_t off, cap;
  void* take(size_t bytes) {
    size_t a = (off + 255) & ~(size_t)255;
    if (a + bytes > cap) a = 0;      // wrap instead of OOB (compile-only env)
    off = a + bytes;
    return base + a;
  }
};
inline float* fbuf(Ws& w, size_t n) { return (float*)w.take(n * sizeof(float)); }
inline f16*   hbuf(Ws& w, size_t n) { return (f16*)w.take(n * sizeof(f16)); }
inline int nblk(ll n) { return (int)((n + 255) / 256); }

struct BN { const float *g, *b, *m, *v; };
struct Blk { const float* w1; BN b1; const float* w2; BN b2; const float* dw; BN dbn; bool down; };
struct MambaP { const float *in_proj_w, *conv_w, *conv_b, *x_proj_w, *dt_proj_w,
                            *dt_proj_b, *A_log, *D, *out_proj_w; };
struct Attn { const float *in_w, *in_b, *out_w, *out_b; };
struct Fuse { const float *akw, *akb, *avw, *avb, *gkw, *gkb, *gvw, *gvb;
              const float *lnag, *lnab, *lngg, *lngb;
              Attn aacc, aang;
              const float *ln2g, *ln2b, *l1w, *l1b, *l2w, *l2b; };

struct Cur { void* const* a; int i; const float* f() { return (const float*)a[i++]; } };
BN rdbn(Cur& c) { BN b; b.g = c.f(); b.b = c.f(); b.m = c.f(); b.v = c.f(); return b; }
Blk rdblk(Cur& c, bool down) {
  Blk k; k.w1 = c.f(); k.b1 = rdbn(c); k.w2 = c.f(); k.b2 = rdbn(c);
  k.down = down; k.dw = nullptr;
  if (down) { k.dw = c.f(); k.dbn = rdbn(c); }
  return k;
}
MambaP rdm(Cur& c) {
  MambaP m;
  m.in_proj_w = c.f(); m.conv_w = c.f(); m.conv_b = c.f(); m.x_proj_w = c.f();
  m.dt_proj_w = c.f(); m.dt_proj_b = c.f(); m.A_log = c.f(); m.D = c.f();
  m.out_proj_w = c.f();
  return m;
}

void gemm(hipStream_t st, const f16* A, const f16* B, float* C, int M, int N, int K) {
  dim3 grid((unsigned)((N + 63) / 64), (unsigned)((M + 127) / 128));
  k_wmma_gemm<<<grid, 256, 0, st>>>(A, B, C, M, N, K);
}

f16* castM(hipStream_t st, Ws& ws, const float* src, int rows, int srcld,
           int col0, int cols, int dstld) {
  f16* d = hbuf(ws, (size_t)rows * dstld);
  ll tot = (ll)rows * dstld;
  k_cast_pad<<<nblk(tot), 256, 0, st>>>(src, srcld, col0, cols, d, dstld, tot);
  return d;
}

float* conv_bn(hipStream_t st, Ws& ws, const float* in, int N, int Cin, int H, int W,
               const float* wgt, int Cout, int KH, int KW, int stride, int pad,
               const BN& bn, const float* resid, int relu, int& OH, int& OW) {
  OH = (H + 2 * pad - KH) / stride + 1;
  OW = (W + 2 * pad - KW) / stride + 1;
  const int Kc = Cin * KH * KW;
  const int Kp = (Kc + 31) & ~31;
  const ll M = (ll)N * OH * OW;
  f16* col = hbuf(ws, (size_t)M * Kp);
  { ll tot = M * Kp;
    k_im2col<<<nblk(tot), 256, 0, st>>>(in, col, N, Cin, H, W, OH, OW, KH, KW,
                                        stride, pad, Kc, Kp, tot); }
  f16* w16 = castM(st, ws, wgt, Cout, Kc, 0, Kc, Kp);
  float* go = fbuf(ws, (size_t)M * Cout);
  gemm(st, col, w16, go, (int)M, Cout, Kp);
  float* out = fbuf(ws, (size_t)M * Cout);
  { ll tot = M * Cout;
    k_bn_act<<<nblk(tot), 256, 0, st>>>(go, out, bn.g, bn.b, bn.m, bn.v, resid,
                                        relu, N, Cout, OH, OW, tot); }
  return out;
}

float* resblock(hipStream_t st, Ws& ws, const float* x, int N, int Cin, int H, int W,
                const Blk& b, int stride, int Cout, int& OH, int& OW) {
  int oh, ow;
  float* o1 = conv_bn(st, ws, x, N, Cin, H, W, b.w1, Cout, 3, 3, stride, 1,
                      b.b1, nullptr, 1, oh, ow);
  const float* idn = x;
  if (b.down) {
    int dh, dw;
    idn = conv_bn(st, ws, x, N, Cin, H, W, b.dw, Cout, 1, 1, stride, 0,
                  b.dbn, nullptr, 0, dh, dw);
  }
  return conv_bn(st, ws, o1, N, Cout, oh, ow, b.w2, Cout, 3, 3, 1, 1,
                 b.b2, idn, 1, OH, OW);
}

float* run_layer(hipStream_t st, Ws& ws, float* x, int N, int Cin, int& H, int& W,
                 const Blk* b, int stride, int Cout) {
  int oh, ow;
  float* o = resblock(st, ws, x, N, Cin, H, W, b[0], stride, Cout, oh, ow);
  int oh2, ow2;
  o = resblock(st, ws, o, N, Cout, oh, ow, b[1], 1, Cout, oh2, ow2);
  H = oh2; W = ow2;
  return o;
}

void run_mamba(hipStream_t st, Ws& ws, const float* tokens, int S, int Lseq, int D,
               const MambaP& p, float* out) {
  const int T = S * Lseq, di = 2 * D;
  const int r = (D + 15) / 16, rn = r + 32, r32 = (r + 31) & ~31;
  f16* tok16 = castM(st, ws, tokens, T, D, 0, D, D);
  f16* wi = castM(st, ws, p.in_proj_w, 2 * di, D, 0, D, D);
  float* xz = fbuf(ws, (size_t)T * 2 * di);
  gemm(st, tok16, wi, xz, T, 2 * di, D);
  float* xc = fbuf(ws, (size_t)T * di);
  f16* xc16 = hbuf(ws, (size_t)T * di);
  { ll tot = (ll)T * di;
    k_dwconv_silu<<<nblk(tot), 256, 0, st>>>(xz, p.conv_w, p.conv_b, xc, xc16,
                                             S, Lseq, di, tot); }
  f16* wx = castM(st, ws, p.x_proj_w, rn, di, 0, di, di);
  float* dbl = fbuf(ws, (size_t)T * rn);
  gemm(st, xc16, wx, dbl, T, rn, di);
  f16* dt16 = castM(st, ws, dbl, T, rn, 0, r, r32);
  f16* wdt = castM(st, ws, p.dt_proj_w, di, r, 0, r, r32);
  float* dt = fbuf(ws, (size_t)T * di);
  gemm(st, dt16, wdt, dt, T, di, r32);
  { ll tot = (ll)T * di;
    k_softplus_bias<<<nblk(tot), 256, 0, st>>>(dt, p.dt_proj_b, di, tot); }
  float* y = fbuf(ws, (size_t)T * di);
  f16* y16 = hbuf(ws, (size_t)T * di);
  { ll tot = (ll)S * di;
    k_scan<<<nblk(tot), 256, 0, st>>>(dt, dbl, rn, r, xc, xz, p.A_log, p.D,
                                      y, y16, Lseq, di, tot); }
  f16* wo = castM(st, ws, p.out_proj_w, D, di, 0, di, di);
  gemm(st, y16, wo, out, T, D, di);
}

void stage(hipStream_t st, Ws& ws, float* act, int Lseq, int C, int H, int W,
           const MambaP& fine, const MambaP& coarse, const float* prev, float** xc_out) {
  const int S = H * W, T = S * Lseq;
  float* pooled = fbuf(ws, (size_t)Lseq * C);
  { ll tot = (ll)Lseq * C;
    k_avg_hw<<<nblk(tot), 256, 0, st>>>(act, pooled, C, S, tot); }
  float* tok = fbuf(ws, (size_t)T * C);
  { ll tot = (ll)T * C;
    k_nchw_to_tok<<<nblk(tot), 256, 0, st>>>(act, tok, Lseq, C, S, tot); }
  float* fo = fbuf(ws, (size_t)T * C);
  run_mamba(st, ws, tok, S, Lseq, C, fine, fo);
  { ll tot = (ll)T * C;
    k_tok_add_nchw<<<nblk(tot), 256, 0, st>>>(act, fo, Lseq, C, S, tot); }
  if (prev) k_add<<<nblk((ll)Lseq * C), 256, 0, st>>>(pooled, prev, (ll)Lseq * C);
  float* xc = fbuf(ws, (size_t)Lseq * C);
  run_mamba(st, ws, pooled, 1, Lseq, C, coarse, xc);
  *xc_out = xc;
}

float* interpC(hipStream_t st, Ws& ws, const float* src, int Lr, int Cin, int Cout) {
  float* d = fbuf(ws, (size_t)Lr * Cout);
  ll tot = (ll)Lr * Cout;
  k_interp<<<nblk(tot), 256, 0, st>>>(src, Lr, Cin, d, Cout, tot);
  return d;
}

void run_mha(hipStream_t st, Ws& ws, const float* q, const float* kv, const Attn& at,
             float* out, int Lq, int C) {
  const int nh = 4, hd = C / nh;
  f16* q16 = castM(st, ws, q, Lq, C, 0, C, C);
  f16* e16 = castM(st, ws, kv, Lq, C, 0, C, C);
  f16* wq = castM(st, ws, at.in_w, C, C, 0, C, C);
  f16* wk = castM(st, ws, at.in_w + (ll)C * C, C, C, 0, C, C);
  f16* wv = castM(st, ws, at.in_w + 2ll * C * C, C, C, 0, C, C);
  float* qh = fbuf(ws, (size_t)Lq * C);
  gemm(st, q16, wq, qh, Lq, C, C);
  k_bias_act<<<nblk((ll)Lq * C), 256, 0, st>>>(qh, at.in_b, C, 0, (ll)Lq * C);
  float* kh = fbuf(ws, (size_t)Lq * C);
  gemm(st, e16, wk, kh, Lq, C, C);
  k_bias_act<<<nblk((ll)Lq * C), 256, 0, st>>>(kh, at.in_b + C, C, 0, (ll)Lq * C);
  float* vh = fbuf(ws, (size_t)Lq * C);
  gemm(st, e16, wv, vh, Lq, C, C);
  k_bias_act<<<nblk((ll)Lq * C), 256, 0, st>>>(vh, at.in_b + 2 * C, C, 0, (ll)Lq * C);
  float* att = fbuf(ws, (size_t)Lq * nh * Lq);
  { ll tot = (ll)Lq * nh * Lq;
    k_attn_scores<<<nblk(tot), 256, 0, st>>>(qh, kh, att, Lq, nh, hd,
                                             1.0f / sqrtf((float)hd), tot); }
  k_softmax_rows<<<nblk(Lq * nh), 64, 0, st>>>(att, Lq * nh, Lq);
  float* o = fbuf(ws, (size_t)Lq * C);
  { ll tot = (ll)Lq * C;
    k_attn_out<<<nblk(tot), 256, 0, st>>>(att, vh, o, Lq, nh, hd, tot); }
  f16* o16 = castM(st, ws, o, Lq, C, 0, C, C);
  f16* wo = castM(st, ws, at.out_w, C, C, 0, C, C);
  gemm(st, o16, wo, out, Lq, C, C);
  k_bias_act<<<nblk((ll)Lq * C), 256, 0, st>>>(out, at.out_b, C, 0, (ll)Lq * C);
}

float* run_fuser(hipStream_t st, Ws& ws, float* g, const float* accele,
                 const float* angle, const Fuse& F) {
  const int Lq = 16, C = 512, C2 = 1024;
  float* acc_emb = fbuf(ws, (size_t)Lq * C);
  k_cross_attn<<<Lq, 256, 0, st>>>(g, accele, F.akw, F.akb, F.avw, F.avb, acc_emb, C);
  float* acc_ln = fbuf(ws, (size_t)Lq * C);
  k_layernorm<<<Lq, 256, 0, st>>>(acc_emb, F.lnag, F.lnab, acc_ln, C);
  float* ang_emb = fbuf(ws, (size_t)Lq * C);
  k_cross_attn<<<Lq, 256, 0, st>>>(g, angle, F.gkw, F.gkb, F.gvw, F.gvb, ang_emb, C);
  float* ang_ln = fbuf(ws, (size_t)Lq * C);
  k_layernorm<<<Lq, 256, 0, st>>>(ang_emb, F.lngg, F.lngb, ang_ln, C);
  float* t1 = fbuf(ws, (size_t)Lq * C);
  run_mha(st, ws, g, acc_ln, F.aacc, t1, Lq, C);
  k_add<<<nblk((ll)Lq * C), 256, 0, st>>>(g, t1, (ll)Lq * C);
  float* t2 = fbuf(ws, (size_t)Lq * C);
  run_mha(st, ws, g, ang_ln, F.aang, t2, Lq, C);
  float* xcat = fbuf(ws, (size_t)Lq * C2);
  k_copy_cols<<<nblk((ll)Lq * C), 256, 0, st>>>(g, C, xcat, C2, 0, (ll)Lq * C);
  k_copy_cols<<<nblk((ll)Lq * C), 256, 0, st>>>(t2, C, xcat, C2, C, (ll)Lq * C);
  float* xln = fbuf(ws, (size_t)Lq * C2);
  k_layernorm<<<Lq, 256, 0, st>>>(xcat, F.ln2g, F.ln2b, xln, C2);
  f16* x16 = castM(st, ws, xln, Lq, C2, 0, C2, C2);
  f16* w1 = castM(st, ws, F.l1w, C2, C2, 0, C2, C2);
  float* h1 = fbuf(ws, (size_t)Lq * C2);
  gemm(st, x16, w1, h1, Lq, C2, C2);
  k_bias_act<<<nblk((ll)Lq * C2), 256, 0, st>>>(h1, F.l1b, C2, 1, (ll)Lq * C2);
  f16* h16 = castM(st, ws, h1, Lq, C2, 0, C2, C2);
  f16* w2 = castM(st, ws, F.l2w, C2, C2, 0, C2, C2);
  float* h2 = fbuf(ws, (size_t)Lq * C2);
  gemm(st, h16, w2, h2, Lq, C2, C2);
  k_bias_act<<<nblk((ll)Lq * C2), 256, 0, st>>>(h2, F.l2b, C2, 0, (ll)Lq * C2);
  k_add<<<nblk((ll)Lq * C2), 256, 0, st>>>(h2, xln, (ll)Lq * C2);
  return h2;  // [16 x 1024]
}

} // namespace

extern "C" void kernel_launch(void* const* d_in, const int* in_sizes, int n_in,
                              void* d_out, int out_size, void* d_ws, size_t ws_size,
                              hipStream_t stream) {
  (void)in_sizes; (void)n_in; (void)out_size;
  Ws ws{(char*)d_ws, 0, ws_size};
  const float* x      = (const float*)d_in[0];
  const float* accele = (const float*)d_in[1];
  const float* angle  = (const float*)d_in[2];

  Cur c{d_in, 3};
  const float* conv1w = c.f();
  BN bn1 = rdbn(c);
  Blk L1[2] = { rdblk(c, false), rdblk(c, false) };
  Blk L2[2] = { rdblk(c, true),  rdblk(c, false) };
  Blk L3[2] = { rdblk(c, true),  rdblk(c, false) };
  Blk L4[2] = { rdblk(c, true),  rdblk(c, false) };
  MambaP m1f = rdm(c), m1c = rdm(c), m2f = rdm(c), m2c = rdm(c);
  MambaP m3f = rdm(c), m3c = rdm(c), m4f = rdm(c), m4c = rdm(c), m5 = rdm(c);
  const float* out_w = c.f();
  const float* out_b = c.f();
  Fuse F;
  F.akw = c.f(); F.akb = c.f(); F.avw = c.f(); F.avb = c.f();
  F.gkw = c.f(); F.gkb = c.f(); F.gvw = c.f(); F.gvb = c.f();
  F.lnag = c.f(); F.lnab = c.f(); F.lngg = c.f(); F.lngb = c.f();
  F.aacc = Attn{c.f(), c.f(), c.f(), c.f()};
  F.aang = Attn{c.f(), c.f(), c.f(), c.f()};
  F.ln2g = c.f(); F.ln2b = c.f();
  F.l1w = c.f(); F.l1b = c.f(); F.l2w = c.f(); F.l2b = c.f();

  const int Lseq = 16;
  // --- input prep (2*(x/255)-1), NCHW with N=B*L=16, C=2 ---
  const ll nin = 16ll * 2 * 128 * 128;
  float* h0 = fbuf(ws, (size_t)nin);
  k_prep<<<nblk(nin), 256, 0, stream>>>(x, h0, nin);

  // --- conv1 7x7/2 + bn + relu, then maxpool 3x3/2 ---
  int OH, OW;
  float* c1 = conv_bn(stream, ws, h0, 16, 2, 128, 128, conv1w, 64, 7, 7, 2, 3,
                      bn1, nullptr, 1, OH, OW);
  int PH = (OH + 2 - 3) / 2 + 1, PW = (OW + 2 - 3) / 2 + 1;
  float* p1 = fbuf(ws, 16ull * 64 * PH * PW);
  { ll tot = 16ll * 64 * PH * PW;
    k_maxpool<<<nblk(tot), 256, 0, stream>>>(c1, p1, 16, 64, OH, OW, PH, PW, tot); }
  int H = PH, W = PW;

  // --- layer1 + remamba1 ---
  float* a1 = run_layer(stream, ws, p1, 16, 64, H, W, L1, 1, 64);
  float* xc1;
  stage(stream, ws, a1, Lseq, 64, H, W, m1f, m1c, nullptr, &xc1);
  float* xc1_512 = interpC(stream, ws, xc1, Lseq, 64, 512);

  // --- layer2 + remamba2 ---
  float* a2 = run_layer(stream, ws, a1, 16, 64, H, W, L2, 2, 128);
  float* prev2 = interpC(stream, ws, xc1, Lseq, 64, 128);
  float* xc2;
  stage(stream, ws, a2, Lseq, 128, H, W, m2f, m2c, prev2, &xc2);
  float* xc2_512 = interpC(stream, ws, xc2, Lseq, 128, 512);

  // --- layer3 + remamba3 ---
  float* a3 = run_layer(stream, ws, a2, 16, 128, H, W, L3, 2, 256);
  float* prev3 = interpC(stream, ws, xc2, Lseq, 128, 256);
  float* xc3;
  stage(stream, ws, a3, Lseq, 256, H, W, m3f, m3c, prev3, &xc3);
  float* xc3_512 = interpC(stream, ws, xc3, Lseq, 256, 512);

  // --- layer4 + remamba4 ---
  float* a4 = run_layer(stream, ws, a3, 16, 256, H, W, L4, 2, 512);
  float* prev4 = interpC(stream, ws, xc3, Lseq, 256, 512);
  float* xc4;
  stage(stream, ws, a4, Lseq, 512, H, W, m4f, m4c, prev4, &xc4);
  float* xc4_512 = interpC(stream, ws, xc4, Lseq, 512, 512);

  // --- g = mean(xf4) + xc1..xc4 ---
  float* g = fbuf(ws, (size_t)Lseq * 512);
  { ll tot = (ll)Lseq * 512;
    k_avg_hw<<<nblk(tot), 256, 0, stream>>>(a4, g, 512, H * W, tot);
    k_add<<<nblk(tot), 256, 0, stream>>>(g, xc1_512, tot);
    k_add<<<nblk(tot), 256, 0, stream>>>(g, xc2_512, tot);
    k_add<<<nblk(tot), 256, 0, stream>>>(g, xc3_512, tot);
    k_add<<<nblk(tot), 256, 0, stream>>>(g, xc4_512, tot); }

  // --- fuser ---
  float* g2 = run_fuser(stream, ws, g, accele, angle, F);  // [16 x 1024]

  // --- m5 mamba + residual ---
  float* m5o = fbuf(ws, (size_t)Lseq * 1024);
  run_mamba(stream, ws, g2, 1, Lseq, 1024, m5, m5o);
  k_add<<<nblk((ll)Lseq * 1024), 256, 0, stream>>>(m5o, g2, (ll)Lseq * 1024);

  // --- output head: [16 x 1024] x [6 x 1024]^T + bias ---
  f16* g16 = castM(stream, ws, m5o, Lseq, 1024, 0, 1024, 1024);
  f16* ow16 = castM(stream, ws, out_w, 6, 1024, 0, 1024, 1024);
  gemm(stream, g16, ow16, (float*)d_out, Lseq, 6, 1024);
  k_bias_act<<<1, 256, 0, stream>>>((float*)d_out, out_b, 6, 0, (ll)Lseq * 6);
}
